// Local_Reconstruction_87462714016293
// MI455X (gfx1250) — matrix operand order repaired and checked by reading, not compiled, not run on hardware
//
#include <hip/hip_runtime.h>
#include <hip/hip_bf16.h>

// ---------------------------------------------------------------------------
// Fused Swin window attention for MI455X (gfx1250, wave32, WMMA).
// One workgroup (128 threads = 4 waves) per window of 64 tokens x 128 dim.
// All GEMMs via v_wmma_f32_16x16x32_bf16 (bf16 operands, f32 accumulate).
// Round 2: register-cache A tiles across N loops; store V transposed so the
// PV B-operand loads are ds_load_b128 instead of 16x ds_load_u16.
// ---------------------------------------------------------------------------

#define NTOK 64          // tokens per window
#define DIM  128         // embedding dim
#define NH   4           // heads
#define HD   32          // head dim
#define QK_SCALE 0.17677669529663687f   // 32^-0.5

typedef __bf16 bf16_t;
typedef __attribute__((ext_vector_type(16))) __bf16 v16bf;
typedef __attribute__((ext_vector_type(8)))  __bf16 v8bf;
typedef __attribute__((ext_vector_type(8)))  float  v8f;
typedef __attribute__((ext_vector_type(4)))  unsigned int u32x4;
typedef __attribute__((ext_vector_type(4)))  float  f32x4;

// LDS arena layout (bytes). Strides padded so row stride mod 64 banks == 4
// (or gcd(stride_words,64) small) -> conflict-free 16-lane row access.
#define XS_LD   136                                  // bf16 elems per row
#define QKV_LD  264                                  // q|k only (cols 0..255)
#define VT_LD   72                                   // V^T [dim][token]
#define PS_LD   72
#define YS_LD   132                                  // f32 elems per row
#define XS_OFF  0
#define QKV_OFF (XS_OFF  + NTOK * XS_LD  * 2)        // 17408
#define VT_OFF  (QKV_OFF + NTOK * QKV_LD * 2)        // 51200  (qkv arena 33792)
#define PS_OFF  (VT_OFF  + DIM  * VT_LD  * 2)        // 69632
#define REL_OFF (PS_OFF  + NH * NTOK * PS_LD * 2)    // 106496
#define BT_OFF  (REL_OFF + NTOK * NTOK)              // 110592
#define SMEM_BYTES (BT_OFF + 225 * NH * 4)           // 114192

__device__ __forceinline__ v8f vzero() {
    v8f z;
#pragma unroll
    for (int i = 0; i < 8; ++i) z[i] = 0.0f;
    return z;
}

__device__ __forceinline__ v8f wmma_bf16(v16bf a, v16bf b, v8f c) {
    return __builtin_amdgcn_wmma_f32_16x16x32_bf16(
        /*neg_a=*/false, a, /*neg_b=*/false, b,
        /*c_mod=*/(short)0, c, /*reuse_a=*/false, /*reuse_b=*/false);
}

// A operand, 16x32 bf16, row-major source in LDS.
// lanes 0-15: row r0+ln, K [c0..c0+15]
// lanes 16-31: same rows, K [c0+16..c0+31]
__device__ __forceinline__ v16bf ld_a(const bf16_t* s, int ld, int r0, int c0,
                                      int hl, int ln) {
    const bf16_t* p = s + (r0 + ln) * ld + c0 + hl * 16;
    v16bf o;
    *(u32x4*)&o       = *(const u32x4*)p;
    *((u32x4*)&o + 1) = *(const u32x4*)(p + 8);
    return o;
}

// B operand 32x16, where B = S^T for a row-major source S (rows = N features).
// lane n holds S-row n0+ln; lanes 0-15 carry K [k0..k0+15], 16-31 [k0+16..k0+31].
__device__ __forceinline__ v16bf ld_b_rows(const bf16_t* s, int ld, int n0,
                                           int k0, int hl, int ln) {
    const bf16_t* p = s + (n0 + ln) * ld + k0 + hl * 16;
    v16bf o;
    *(u32x4*)&o       = *(const u32x4*)p;
    *((u32x4*)&o + 1) = *(const u32x4*)(p + 8);
    return o;
}

// Same as ld_b_rows but from a row-major f32 global weight matrix (convert).
__device__ __forceinline__ v16bf ld_b_rows_f32(const float* s, int ld, int n0,
                                               int k0, int hl, int ln) {
    const float* p = s + (n0 + ln) * ld + k0 + hl * 16;
    v16bf o;
#pragma unroll
    for (int i = 0; i < 16; ++i) o[i] = (bf16_t)p[i];
    return o;
}

__global__ __launch_bounds__(128)
void win_attn_fused_kernel(const float* __restrict__ x,
                           const float* __restrict__ qkv_w,
                           const float* __restrict__ qkv_b,
                           const float* __restrict__ proj_w,
                           const float* __restrict__ proj_b,
                           const float* __restrict__ bias_table,
                           const int*   __restrict__ rel_index,
                           float* __restrict__ out) {
    extern __shared__ unsigned char smem[];
    bf16_t* xs  = (bf16_t*)(smem + XS_OFF);    // x tile (later: attention out O)
    bf16_t* qkv = (bf16_t*)(smem + QKV_OFF);   // [64][256] q|k (later: ys f32)
    bf16_t* vt  = (bf16_t*)(smem + VT_OFF);    // V^T [128 dims][64 tokens]
    bf16_t* ps  = (bf16_t*)(smem + PS_OFF);    // per-head softmax probs
    unsigned char* rel = smem + REL_OFF;       // rel_index as u8
    float*  btab = (float*)(smem + BT_OFF);    // bias table [225][4]
    float*  ys   = (float*)(smem + QKV_OFF);   // reuse q|k arena for f32 out

    const int tid  = threadIdx.x;
    const int lane = tid & 31;
    const int wv   = tid >> 5;     // wave id 0..3
    const int hl   = lane >> 4;    // half-wave
    const int ln   = lane & 15;
    const int b    = blockIdx.x;

    // ---- Phase 0: stage x (f32 -> bf16), rel_index (i32 -> u8), bias table.
    // Also prefetch the (L2-resident) weight matrices.
    {
        for (int off = tid * 256; off < 3 * DIM * DIM * 4; off += 128 * 256)
            __builtin_prefetch((const char*)qkv_w + off, 0, 2);
        for (int off = tid * 256; off < DIM * DIM * 4; off += 128 * 256)
            __builtin_prefetch((const char*)proj_w + off, 0, 2);

        const f32x4* xg = (const f32x4*)(x + (size_t)b * (NTOK * DIM));
        for (int v = tid; v < NTOK * DIM / 4; v += 128) {
            f32x4 f = xg[v];
            int r = v >> 5;
            int c = (v & 31) << 2;
            bf16_t* p = xs + r * XS_LD + c;
            p[0] = (bf16_t)f.x; p[1] = (bf16_t)f.y;
            p[2] = (bf16_t)f.z; p[3] = (bf16_t)f.w;
        }
        for (int v = tid; v < NTOK * NTOK; v += 128)
            rel[v] = (unsigned char)rel_index[v];
        for (int v = tid; v < 225 * NH; v += 128)
            btab[v] = bias_table[v];
    }
    __syncthreads();

    // ---- Phase 1: QKV = x @ qkv_w^T + qkv_b   (M=64, N=384, K=128)
    // Each wave owns 96 output features (6 N-tiles). All 16 X A-tiles are
    // cached in registers (128 VGPRs) and reused across the 6 N-tiles.
    {
        v16bf Ax[16];
#pragma unroll
        for (int kt = 0; kt < 4; ++kt)
#pragma unroll
            for (int m = 0; m < 4; ++m)
                Ax[4 * kt + m] = ld_a(xs, XS_LD, m * 16, kt * 32, hl, ln);

        const int n0w = wv * 96;
        for (int nt = 0; nt < 6; ++nt) {
            const int n0 = n0w + nt * 16;
            v8f acc[4];
#pragma unroll
            for (int m = 0; m < 4; ++m) acc[m] = vzero();
#pragma unroll
            for (int kt = 0; kt < 4; ++kt) {
                v16bf Bt = ld_b_rows_f32(qkv_w, DIM, n0, kt * 32, hl, ln);
#pragma unroll
                for (int m = 0; m < 4; ++m)
                    acc[m] = wmma_bf16(Ax[4 * kt + m], Bt, acc[m]);
            }
            const int ng = n0 + ln;
            const float bb = qkv_b[ng];
            if (ng < 2 * DIM) {
                // q (pre-scaled) and k, row-major [token][feature]
                const float sc = (ng < DIM) ? QK_SCALE : 1.0f;
#pragma unroll
                for (int m = 0; m < 4; ++m)
#pragma unroll
                    for (int i = 0; i < 8; ++i) {
                        int r = m * 16 + hl * 8 + i;
                        qkv[r * QKV_LD + ng] = (bf16_t)((acc[m][i] + bb) * sc);
                    }
            } else {
                // v stored transposed [dim][token]: 8 contiguous bf16 / lane
                bf16_t* vp = vt + (ng - 2 * DIM) * VT_LD;
#pragma unroll
                for (int m = 0; m < 4; ++m) {
                    v8bf pk;
#pragma unroll
                    for (int i = 0; i < 8; ++i)
                        pk[i] = (bf16_t)(acc[m][i] + bb);
                    *(v8bf*)(vp + m * 16 + hl * 8) = pk;
                }
            }
        }
    }
    __syncthreads();

    // ---- Phase 2: attention, one head per wave
    {
        const int h  = wv;
        const int qc = h * HD;
        const int kc = DIM + h * HD;
        bf16_t* psh = ps + h * (NTOK * PS_LD);

        v16bf Bk[4];
#pragma unroll
        for (int nt = 0; nt < 4; ++nt)
            Bk[nt] = ld_b_rows(qkv, QKV_LD, nt * 16, kc, hl, ln);

        for (int m = 0; m < 4; ++m) {
            v16bf Aq = ld_a(qkv, QKV_LD, m * 16, qc, hl, ln);
            v8f s[4];
#pragma unroll
            for (int nt = 0; nt < 4; ++nt)
                s[nt] = wmma_bf16(Aq, Bk[nt], vzero());

            // relative position bias gather-add
#pragma unroll
            for (int nt = 0; nt < 4; ++nt) {
                int col = nt * 16 + ln;
#pragma unroll
                for (int i = 0; i < 8; ++i) {
                    int row = m * 16 + hl * 8 + i;
                    s[nt][i] += btab[(int)rel[row * NTOK + col] * NH + h];
                }
            }

            // row softmax: each 16-lane half owns full rows across 4 n-tiles
            float rmax[8], rinv[8];
#pragma unroll
            for (int i = 0; i < 8; ++i) {
                float v = fmaxf(fmaxf(s[0][i], s[1][i]),
                                fmaxf(s[2][i], s[3][i]));
                v = fmaxf(v, __shfl_xor(v, 1, 32));
                v = fmaxf(v, __shfl_xor(v, 2, 32));
                v = fmaxf(v, __shfl_xor(v, 4, 32));
                v = fmaxf(v, __shfl_xor(v, 8, 32));
                rmax[i] = v;
            }
#pragma unroll
            for (int i = 0; i < 8; ++i) {
                float t = 0.0f;
#pragma unroll
                for (int nt = 0; nt < 4; ++nt) {
                    s[nt][i] = __expf(s[nt][i] - rmax[i]);
                    t += s[nt][i];
                }
                t += __shfl_xor(t, 1, 32);
                t += __shfl_xor(t, 2, 32);
                t += __shfl_xor(t, 4, 32);
                t += __shfl_xor(t, 8, 32);
                rinv[i] = 1.0f / t;
            }
            // store normalized probabilities (bf16) for the PV GEMM
#pragma unroll
            for (int nt = 0; nt < 4; ++nt) {
                int col = nt * 16 + ln;
#pragma unroll
                for (int i = 0; i < 8; ++i) {
                    int row = m * 16 + hl * 8 + i;
                    psh[row * PS_LD + col] = (bf16_t)(s[nt][i] * rinv[i]);
                }
            }
        }

        // O = P @ V   (M=64, N=32, K=64) -> write into xs arena (x dead now)
        // B operand from V^T rows: pure ds_load_b128.
        v16bf Bv[2][2];
#pragma unroll
        for (int kt = 0; kt < 2; ++kt)
#pragma unroll
            for (int nt = 0; nt < 2; ++nt)
                Bv[kt][nt] = ld_b_rows(vt, VT_LD, h * HD + nt * 16, kt * 32,
                                       hl, ln);

        for (int m = 0; m < 4; ++m) {
            v8f o[2] = {vzero(), vzero()};
#pragma unroll
            for (int kt = 0; kt < 2; ++kt) {
                v16bf Ap = ld_a(psh, PS_LD, m * 16, kt * 32, hl, ln);
#pragma unroll
                for (int nt = 0; nt < 2; ++nt)
                    o[nt] = wmma_bf16(Ap, Bv[kt][nt], o[nt]);
            }
#pragma unroll
            for (int nt = 0; nt < 2; ++nt) {
                int col = h * HD + nt * 16 + ln;
#pragma unroll
                for (int i = 0; i < 8; ++i) {
                    int row = m * 16 + hl * 8 + i;
                    xs[row * XS_LD + col] = (bf16_t)o[nt][i];
                }
            }
        }
    }
    __syncthreads();

    // ---- Phase 3: Y = O @ proj_w^T + proj_b  (M=64, N=128, K=128)
    // Each wave owns 32 output features; O A-tiles register-cached.
    {
        v16bf Ao[16];
#pragma unroll
        for (int kt = 0; kt < 4; ++kt)
#pragma unroll
            for (int m = 0; m < 4; ++m)
                Ao[4 * kt + m] = ld_a(xs, XS_LD, m * 16, kt * 32, hl, ln);

        const int n0w = wv * 32;
#pragma unroll
        for (int nt = 0; nt < 2; ++nt) {
            const int n0 = n0w + nt * 16;
            v8f acc[4];
#pragma unroll
            for (int m = 0; m < 4; ++m) acc[m] = vzero();
#pragma unroll
            for (int kt = 0; kt < 4; ++kt) {
                v16bf Bt = ld_b_rows_f32(proj_w, DIM, n0, kt * 32, hl, ln);
#pragma unroll
                for (int m = 0; m < 4; ++m)
                    acc[m] = wmma_bf16(Ao[4 * kt + m], Bt, acc[m]);
            }
            const float bb = proj_b[n0 + ln];
#pragma unroll
            for (int m = 0; m < 4; ++m)
#pragma unroll
                for (int i = 0; i < 8; ++i) {
                    int row = m * 16 + hl * 8 + i;
                    ys[row * YS_LD + n0 + ln] = acc[m][i] + bb;
                }
        }
    }
    __syncthreads();

    // ---- Phase 4: coalesced f32 store to global
    {
        f32x4* og = (f32x4*)(out + (size_t)b * (NTOK * DIM));
        for (int v = tid; v < NTOK * DIM / 4; v += 128) {
            int r = v >> 5;
            int c = (v & 31) << 2;
            og[v] = *(const f32x4*)(ys + r * YS_LD + c);
        }
    }
}

extern "C" void kernel_launch(void* const* d_in, const int* in_sizes, int n_in,
                              void* d_out, int out_size, void* d_ws, size_t ws_size,
                              hipStream_t stream) {
    (void)n_in; (void)out_size; (void)d_ws; (void)ws_size;
    const float* x      = (const float*)d_in[0];
    const float* qkv_w  = (const float*)d_in[1];
    const float* qkv_b  = (const float*)d_in[2];
    const float* proj_w = (const float*)d_in[3];
    const float* proj_b = (const float*)d_in[4];
    const float* btab   = (const float*)d_in[5];
    const int*   rel    = (const int*)d_in[6];
    float* out = (float*)d_out;

    const int B = in_sizes[0] / (NTOK * DIM);   // number of windows (8192)

    // allow >64KB dynamic LDS (gfx1250 WGP has 320KB)
    (void)hipFuncSetAttribute((const void*)win_attn_fused_kernel,
                              hipFuncAttributeMaxDynamicSharedMemorySize,
                              SMEM_BYTES);

    win_attn_fused_kernel<<<B, 128, SMEM_BYTES, stream>>>(
        x, qkv_w, qkv_b, proj_w, proj_b, btab, rel, out);
}
